// LSTMModel_82274393522269
// MI455X (gfx1250) — matrix operand order repaired and checked by reading, not compiled, not run on hardware
//
#include <hip/hip_runtime.h>
#include <hip/hip_bf16.h>

// ---------------------------------------------------------------------------
// Fused 2-layer LSTM scan + FC for gfx1250 (MI455X).
//   grid = 16 workgroups (one per 16-row batch slice), block = 512 (16 waves).
//   Wave pair per hidden 16-col group: even wave -> gates i,g ; odd -> f,o.
//   GEMMs via v_wmma_f32_16x16x32_bf16; weights resident in LDS / VGPRs.
//   K-chunk fragment loads are software-pipelined 1 chunk ahead.
// ---------------------------------------------------------------------------

#define TT 2048
#define BB 256
#define II 32
#define HH 128
#define G4 512   // 4*H

typedef __attribute__((ext_vector_type(16))) __bf16       v16bf;
typedef __attribute__((ext_vector_type(8)))  float        v8f;
typedef __attribute__((ext_vector_type(4)))  unsigned int uv4;

union Frag {
  v16bf v;
  uv4   q[2];
  unsigned short s[16];
};

__device__ __forceinline__ unsigned short f2bf(float f) {
  unsigned int u = __float_as_uint(f);
  u += 0x7FFFu + ((u >> 16) & 1u);            // round-to-nearest-even
  return (unsigned short)(u >> 16);
}
__device__ __forceinline__ float sigm(float x) { return 1.f / (1.f + __expf(-x)); }
__device__ __forceinline__ float tanhx(float x) {
  float e = __expf(-2.f * fabsf(x));
  float t = (1.f - e) / (1.f + e);
  return copysignf(t, x);
}

// A-fragment (16x32 bf16, row-major block, row stride rs elements).
// lane<16 : row=lane,    K = kb+[0..15]
// lane>=16: row=lane-16, K = kb+[16..31]
__device__ __forceinline__ Frag ldsA(const unsigned short* base, int row, int rs,
                                     int kb, int khalf) {
  const unsigned short* p = base + row * rs + kb + khalf * 16;
  Frag f;
  f.q[0] = *(const uv4*)p;
  f.q[1] = *(const uv4*)(p + 8);
  return f;
}

// B-fragment (32x16 bf16). Weight rows stored per output column (gate), rs wide.
// lane: col = lane%16, K = kb + (lane/16)*16 + [0..15]  (32 contiguous bytes)
__device__ __forceinline__ Frag ldsB(const unsigned short* base, int col, int rs,
                                     int kb, int khalf) {
  const unsigned short* p = base + col * rs + kb + khalf * 16;
  Frag f;
  f.q[0] = *(const uv4*)p;
  f.q[1] = *(const uv4*)(p + 8);
  return f;
}

// 16 consecutive f32 from global -> bf16 B-fragment half (kept in VGPRs).
__device__ __forceinline__ Frag gfrag(const float* p) {
  Frag f;
#pragma unroll
  for (int i = 0; i < 16; ++i) f.s[i] = f2bf(p[i]);
  return f;
}

__device__ __forceinline__ v8f wmma_bf16(v16bf a, v16bf b, v8f c) {
  return __builtin_amdgcn_wmma_f32_16x16x32_bf16(false, a, false, b, (short)0, c,
                                                 false, false);
}

// LDS layout (bytes), total 322048 <= 327680 (320 KiB)
#define OFF_W0  0                      // [512][160] bf16  (W_ih0 | W_hh0)
#define OFF_W1  163840                 // [512][128] bf16  (W_ih1)
#define OFF_X   294912                 // 2 x [16][32] bf16  x_t double buffer
#define OFF_H0  296960                 // [16][128] bf16   h0 state
#define OFF_H1  301056                 // [16][128] bf16   h1 state
#define OFF_FC  305152                 // [8][16]   f32    FC partials
#define OFF_GEX 305664                 // [8][2][16][16] f32 gate exchange
#define SMEM_BYTES 322048

__global__ void lstm2_fused_kernel(const float* __restrict__ x,
                                   const float* __restrict__ Wih0,
                                   const float* __restrict__ Whh0,
                                   const float* __restrict__ bih0,
                                   const float* __restrict__ bhh0,
                                   const float* __restrict__ Wih1,
                                   const float* __restrict__ Whh1,
                                   const float* __restrict__ bih1,
                                   const float* __restrict__ bhh1,
                                   const float* __restrict__ Wfc,
                                   const float* __restrict__ bfc,
                                   float* __restrict__ out) {
  extern __shared__ char smem[];
  unsigned short* w0lds = (unsigned short*)(smem + OFF_W0);
  unsigned short* w1lds = (unsigned short*)(smem + OFF_W1);
  unsigned short* xblk  = (unsigned short*)(smem + OFF_X);   // 2 buffers of 512
  unsigned short* h0blk = (unsigned short*)(smem + OFF_H0);
  unsigned short* h1blk = (unsigned short*)(smem + OFF_H1);
  float*          fcp   = (float*)(smem + OFF_FC);
  float*          gex   = (float*)(smem + OFF_GEX);

  const int tid   = threadIdx.x;
  const int wv    = tid >> 5;        // wave 0..15
  const int lane  = tid & 31;
  const int lcol  = lane & 15;       // N / column within tile
  const int khalf = lane >> 4;
  const int grp   = wv >> 1;         // hidden col group 0..7 (16 cols each)
  const int pw    = wv & 1;          // 0: gates i,g   1: gates f,o
  const int hb    = grp * 16;
  const int bbase = blockIdx.x * 16; // batch slice

  // gate-output columns owned by this wave (B-fragment col index incl. lane)
  int gcol[2];
  gcol[0] = (pw)     * HH + hb + lcol;   // i or f
  gcol[1] = (2 + pw) * HH + hb + lcol;   // g or o

  // ---- one-time: pack weights into LDS as bf16 ----
  for (int g = tid; g < G4; g += 512) {
    const float* ih = Wih0 + g * II;
    const float* hh = Whh0 + g * HH;
    unsigned short* d0 = w0lds + g * 160;
#pragma unroll 4
    for (int i = 0; i < II; ++i) d0[i] = f2bf(ih[i]);
#pragma unroll 4
    for (int k = 0; k < HH; ++k) d0[II + k] = f2bf(hh[k]);
    const float* ih1 = Wih1 + g * HH;
    unsigned short* d1 = w1lds + g * HH;
#pragma unroll 4
    for (int k = 0; k < HH; ++k) d1[k] = f2bf(ih1[k]);
  }
  // zero h states
  for (int i = tid; i < 16 * HH; i += 512) { h0blk[i] = 0; h1blk[i] = 0; }

  // ---- W_hh1 resident in VGPRs: 2 gate tiles x 4 K-chunks per wave ----
  Frag w1hh[2][4];
#pragma unroll
  for (int qi = 0; qi < 2; ++qi)
#pragma unroll
    for (int kc = 0; kc < 4; ++kc)
      w1hh[qi][kc] = gfrag(Whh1 + (size_t)gcol[qi] * HH + kc * 32 + khalf * 16);

  // per-lane gate biases (depend only on gate column N)
  float bs0[2], bs1[2];
#pragma unroll
  for (int qi = 0; qi < 2; ++qi) {
    bs0[qi] = bih0[gcol[qi]] + bhh0[gcol[qi]];
    bs1[qi] = bih1[gcol[qi]] + bhh1[gcol[qi]];
  }
  const float wfc  = Wfc[hb + lcol];
  const float bfc0 = bfc[0];

  // x staging: each thread owns 1 f32 of the 16x32 x_t slice, double-buffered
  const int xrow = tid >> 5;        // rows 0..15
  const int xcl  = tid & 31;
  const float* xsrc = x + ((size_t)bbase + xrow) * II + xcl;
  xblk[xrow * II + xcl] = f2bf(*xsrc);                // buffer 0 <- x_0
  float xreg = (TT > 1) ? xsrc[(size_t)BB * II] : 0.f; // x_1 in flight

  float c0[8], c1[8];
#pragma unroll
  for (int r = 0; r < 8; ++r) { c0[r] = 0.f; c1[r] = 0.f; }

  float* gexg = gex + grp * 512;    // this group's exchange (2 tiles x 256 f32)

  __syncthreads();

#pragma unroll 1
  for (int t = 0; t < TT; ++t) {
    const unsigned short* xcur = xblk + (t & 1) * 512;

    // ================= layer 0 : gates = [x_t | h0] @ W0cat^T + b =========
    v8f a0[2];
#pragma unroll
    for (int qi = 0; qi < 2; ++qi)
#pragma unroll
      for (int r = 0; r < 8; ++r) a0[qi][r] = bs0[qi];

    {
      Frag fa[2], fb[2][2];
      fa[0]    = ldsA(xcur, lcol, II, 0, khalf);
      fb[0][0] = ldsB(w0lds, gcol[0], 160, 0, khalf);
      fb[0][1] = ldsB(w0lds, gcol[1], 160, 0, khalf);
#pragma unroll
      for (int kc = 0; kc < 5; ++kc) {
        const int cur = kc & 1, nxt = cur ^ 1;
        if (kc + 1 < 5) {   // prefetch chunk kc+1 before consuming chunk kc
          fa[nxt]    = ldsA(h0blk, lcol, HH, kc * 32, khalf);
          fb[nxt][0] = ldsB(w0lds, gcol[0], 160, (kc + 1) * 32, khalf);
          fb[nxt][1] = ldsB(w0lds, gcol[1], 160, (kc + 1) * 32, khalf);
        }
#pragma unroll
        for (int qi = 0; qi < 2; ++qi)
          a0[qi] = wmma_bf16(fa[cur].v, fb[cur][qi].v, a0[qi]);
      }
    }

    // odd wave publishes raw f,o gate tiles
    if (pw) {
#pragma unroll
      for (int qi = 0; qi < 2; ++qi)
#pragma unroll
        for (int r = 0; r < 8; ++r)
          gexg[qi * 256 + (r + khalf * 8) * 16 + lcol] = a0[qi][r];
    }
    __syncthreads();                  // (1) all h0 reads + gex writes done

    if (!pw) {                        // even wave: activations + state update
      float h0v[8];
#pragma unroll
      for (int r = 0; r < 8; ++r) {
        int idx  = (r + khalf * 8) * 16 + lcol;
        float ig = sigm(a0[0][r]);
        float gg = tanhx(a0[1][r]);
        float fg = sigm(gexg[idx]);
        float og = sigm(gexg[256 + idx]);
        c0[r]  = fg * c0[r] + ig * gg;
        h0v[r] = og * tanhx(c0[r]);
        h0blk[(r + khalf * 8) * HH + hb + lcol] = f2bf(h0v[r]);
      }
    }
    __syncthreads();                  // (2) h0_t visible

    // ================= layer 1 : gates = [h0_t | h1] @ W1cat^T + b ========
    v8f a1[2];
#pragma unroll
    for (int qi = 0; qi < 2; ++qi)
#pragma unroll
      for (int r = 0; r < 8; ++r) a1[qi][r] = bs1[qi];

    {
      Frag fa[2], fb[2][2];
      fa[0]    = ldsA(h0blk, lcol, HH, 0, khalf);
      fb[0][0] = ldsB(w1lds, gcol[0], HH, 0, khalf);
      fb[0][1] = ldsB(w1lds, gcol[1], HH, 0, khalf);
#pragma unroll
      for (int kc = 0; kc < 8; ++kc) {
        const int cur = kc & 1, nxt = cur ^ 1;
        if (kc + 1 < 8) {   // prefetch chunk kc+1
          const int kn = kc + 1;
          fa[nxt] = (kn < 4) ? ldsA(h0blk, lcol, HH, kn * 32, khalf)
                             : ldsA(h1blk, lcol, HH, (kn - 4) * 32, khalf);
          if (kn < 4) {
            fb[nxt][0] = ldsB(w1lds, gcol[0], HH, kn * 32, khalf);
            fb[nxt][1] = ldsB(w1lds, gcol[1], HH, kn * 32, khalf);
          } else {
            fb[nxt][0] = w1hh[0][kn - 4];
            fb[nxt][1] = w1hh[1][kn - 4];
          }
        }
#pragma unroll
        for (int qi = 0; qi < 2; ++qi)
          a1[qi] = wmma_bf16(fa[cur].v, fb[cur][qi].v, a1[qi]);
      }
    }

    if (pw) {
#pragma unroll
      for (int qi = 0; qi < 2; ++qi)
#pragma unroll
        for (int r = 0; r < 8; ++r)
          gexg[qi * 256 + (r + khalf * 8) * 16 + lcol] = a1[qi][r];
    }
    __syncthreads();                  // (3) all h1 reads + gex writes done

    if (!pw) {
      float h1v[8];
      float p[8];
#pragma unroll
      for (int r = 0; r < 8; ++r) {
        int idx  = (r + khalf * 8) * 16 + lcol;
        float ig = sigm(a1[0][r]);
        float gg = tanhx(a1[1][r]);
        float fg = sigm(gexg[idx]);
        float og = sigm(gexg[256 + idx]);
        c1[r]  = fg * c1[r] + ig * gg;
        h1v[r] = og * tanhx(c1[r]);
        h1blk[(r + khalf * 8) * HH + hb + lcol] = f2bf(h1v[r]);
        p[r] = fmaxf(h1v[r], 0.f) * wfc;
      }
      // FC partial: reduce over this group's 16 columns (16-lane butterfly)
#pragma unroll
      for (int m = 1; m < 16; m <<= 1)
#pragma unroll
        for (int r = 0; r < 8; ++r) p[r] += __shfl_xor(p[r], m, 16);
      if (lcol == 0) {
#pragma unroll
        for (int r = 0; r < 8; ++r) fcp[grp * 16 + khalf * 8 + r] = p[r];
      }
    }

    // stage x_{t+1} into the alternate buffer; fetch x_{t+2}
    if (t + 1 < TT) {
      xblk[(~t & 1) * 512 + xrow * II + xcl] = f2bf(xreg);
      if (t + 2 < TT) xreg = xsrc[(size_t)(t + 2) * BB * II];
    }
    __syncthreads();                  // (4) h1_t, fc partials, x_{t+1} visible

    if (tid < 16) {                   // finalize 16 outputs for this step
      float s = bfc0;
#pragma unroll
      for (int g2 = 0; g2 < 8; ++g2) s += fcp[g2 * 16 + tid];
      out[(size_t)t * BB + bbase + tid] = s;
    }
  }
}

extern "C" void kernel_launch(void* const* d_in, const int* in_sizes, int n_in,
                              void* d_out, int out_size, void* d_ws, size_t ws_size,
                              hipStream_t stream) {
  (void)in_sizes; (void)n_in; (void)out_size; (void)d_ws; (void)ws_size;
  const float* x    = (const float*)d_in[0];
  const float* Wih0 = (const float*)d_in[1];
  const float* Whh0 = (const float*)d_in[2];
  const float* bih0 = (const float*)d_in[3];
  const float* bhh0 = (const float*)d_in[4];
  const float* Wih1 = (const float*)d_in[5];
  const float* Whh1 = (const float*)d_in[6];
  const float* bih1 = (const float*)d_in[7];
  const float* bhh1 = (const float*)d_in[8];
  const float* Wfc  = (const float*)d_in[9];
  const float* bfc  = (const float*)d_in[10];
  float* out = (float*)d_out;

  dim3 grid(BB / 16);   // 16 workgroups, one per batch slice
  dim3 block(512);      // 16 waves: wave pair per hidden col group
  lstm2_fused_kernel<<<grid, block, SMEM_BYTES, stream>>>(
      x, Wih0, Whh0, bih0, bhh0, Wih1, Whh1, bih1, bhh1, Wfc, bfc, out);
}
